// HGraphConvWindow_3143916060813
// MI455X (gfx1250) — compile-verified, matrix operand order checked by reading, not run on hardware
//
#include <hip/hip_runtime.h>

// ---------------------------------------------------------------------------
// Types for CDNA5 WMMA (wave32): v_wmma_f32_16x16x32_bf16
// ---------------------------------------------------------------------------
typedef __attribute__((ext_vector_type(16))) __bf16 v16bf;
typedef __attribute__((ext_vector_type(8)))  float  v8f;

union FragBF {                 // 16 bf16 = 32 bytes = two 16B memory chunks
    v16bf v;
    uint4 q[2];
};

__device__ __forceinline__ unsigned short f2bf(float f) {
    // round-to-nearest-even f32 -> bf16
    unsigned int u = __float_as_uint(f);
    unsigned int r = u + 0x7FFFu + ((u >> 16) & 1u);
    return (unsigned short)(r >> 16);
}

// ---------------------------------------------------------------------------
// 1) Degree accumulation (segment_sum of ones) + rsqrt(clip(deg,1))
// ---------------------------------------------------------------------------
__global__ __launch_bounds__(256) void deg_count_kernel(
    const int* __restrict__ src, const int* __restrict__ dst,
    float* __restrict__ deg_out, float* __restrict__ deg_in, int E)
{
    int e = blockIdx.x * blockDim.x + threadIdx.x;
    if (e >= E) return;
    atomicAdd(&deg_out[src[e]], 1.0f);
    atomicAdd(&deg_in [dst[e]], 1.0f);
}

__global__ __launch_bounds__(256) void deg_rsqrt_kernel(float* __restrict__ deg, int n)
{
    int i = blockIdx.x * blockDim.x + threadIdx.x;
    if (i >= n) return;
    deg[i] = rsqrtf(fmaxf(deg[i], 1.0f));
}

// ---------------------------------------------------------------------------
// 2) Edge-parallel scatter-add:  agg[dst] += x[src] * outdeg_scale[src]
//    Each thread moves 16 floats (4 x float4) of one edge's [T,F] payload.
// ---------------------------------------------------------------------------
__global__ __launch_bounds__(256) void scatter_add_kernel(
    const float* __restrict__ x,          // [N_src, 2048]
    const int* __restrict__ src, const int* __restrict__ dst,
    const float* __restrict__ out_scale,  // [N_src] = outdeg^-1/2
    float* __restrict__ agg,              // [N_dst, 2048]
    int E)
{
    const int CH = 128;                   // 2048 floats / 16 per thread
    long long tid = (long long)blockIdx.x * blockDim.x + threadIdx.x;
    if (tid >= (long long)E * CH) return;
    int e = (int)(tid >> 7);
    int c = (int)(tid & 127);
    int s = src[e], d = dst[e];
    float sc = out_scale[s];
    const float4* xp = (const float4*)(x + (size_t)s * 2048 + c * 16);
    float* ap = agg + (size_t)d * 2048 + c * 16;
#pragma unroll
    for (int i = 0; i < 4; ++i) {
        float4 v = xp[i];
        atomicAdd(ap + 4 * i + 0, v.x * sc);
        atomicAdd(ap + 4 * i + 1, v.y * sc);
        atomicAdd(ap + 4 * i + 2, v.z * sc);
        atomicAdd(ap + 4 * i + 3, v.w * sc);
    }
}

// ---------------------------------------------------------------------------
// 3) Generic f32 -> bf16 conversion
// ---------------------------------------------------------------------------
__global__ __launch_bounds__(256) void f32_to_bf16_kernel(
    const float* __restrict__ src, unsigned short* __restrict__ dst, int n)
{
    int i = blockIdx.x * blockDim.x + threadIdx.x;
    if (i >= n) return;
    dst[i] = f2bf(src[i]);
}

// 4) Weight transpose  w[T,F,H] (f32) -> wT[T,H,F] (bf16)   T=32,F=64,H=128
__global__ __launch_bounds__(256) void w_transpose_bf16_kernel(
    const float* __restrict__ w, unsigned short* __restrict__ wT)
{
    int i = blockIdx.x * blockDim.x + threadIdx.x;   // over 32*128*64
    if (i >= 32 * 128 * 64) return;
    int f = i & 63;
    int h = (i >> 6) & 127;
    int t = i >> 13;
    wT[i] = f2bf(w[((size_t)t * 64 + f) * 128 + h]);
}

// ---------------------------------------------------------------------------
// 5) Fused GraphConv GEMM + epilogue (bf16 WMMA).
//    Block = (timestep t, 4 adjacent M-tiles). The 16KB weight tile wT[t] is
//    staged once per block into LDS via the CDNA5 async global->LDS path
//    (ASYNCcnt-tracked), then all 4 waves consume it with ds_load_b128.
//    h = (indeg^-1/2 * (agg @ w)) + b ;  s = leaky(h) . lw + lb
// ---------------------------------------------------------------------------
#define WROW 72   // padded LDS row stride (shorts): 144B, 16B aligned, bank-skewed

__global__ __launch_bounds__(128) void graph_gemm_fused_kernel(
    const unsigned short* __restrict__ aggbf,   // [Nd, 32, 64] bf16
    const unsigned short* __restrict__ wTbf,    // [32, 128, 64] bf16
    const float* __restrict__ bias,             // [32, 128]
    const float* __restrict__ in_scale,         // [Nd]
    const float* __restrict__ lw,               // [32, 128]
    const float* __restrict__ lb,               // [32]
    float* __restrict__ s_out,                  // [Nd, 32]
    int Nd, int ntiles)
{
    __shared__ unsigned short smemW[128 * WROW];   // 18KB weight tile for this t

    int t  = blockIdx.x & 31;          // timestep
    int tg = blockIdx.x >> 5;          // tile group (4 tiles)

    // ---- async global->LDS stage of wT[t] (128 rows x 128B) ----
    {
        int row = threadIdx.x;         // 128 threads == 128 rows
        unsigned lds = (unsigned)(size_t)(&smemW[row * WROW]);
        unsigned long long g =
            (unsigned long long)(wTbf + ((size_t)t * 128 + row) * 64);
#pragma unroll
        for (int j = 0; j < 8; ++j) {  // 8 x 16B per row
            asm volatile("global_load_async_to_lds_b128 %0, %1, off"
                         :: "v"(lds + j * 16), "v"(g + (unsigned long long)(j * 16))
                         : "memory");
        }
        asm volatile("s_wait_asynccnt 0" ::: "memory");
    }
    __syncthreads();

    int tile = tg * 4 + (threadIdx.x >> 5);
    if (tile >= ntiles) return;
    int m0   = tile * 16;
    int lane = threadIdx.x & 31;
    int half = lane >> 4;              // K-half selector (CDNA5 A/B layout)
    int l16  = lane & 15;
    int kbase = half * 8;

    int arow = m0 + l16; if (arow >= Nd) arow = Nd - 1;   // clamp reads
    const unsigned short* abase = aggbf + ((size_t)arow * 32 + t) * 64;
    FragBF a0, a1;
    a0.q[0] = *(const uint4*)(abase + 0  + kbase);   // K 0..7  / 8..15
    a0.q[1] = *(const uint4*)(abase + 16 + kbase);   // K 16..23 / 24..31
    a1.q[0] = *(const uint4*)(abase + 32 + kbase);
    a1.q[1] = *(const uint4*)(abase + 48 + kbase);

    float part[8];
#pragma unroll
    for (int r = 0; r < 8; ++r) part[r] = 0.0f;

    float rsc[8];
#pragma unroll
    for (int r = 0; r < 8; ++r) {
        int m = m0 + half * 8 + r;
        rsc[r] = in_scale[m < Nd ? m : (Nd - 1)];
    }

    for (int nb = 0; nb < 8; ++nb) {
        int h = nb * 16 + l16;
        const unsigned short* bbase = &smemW[h * WROW];
        FragBF b0, b1;
        b0.q[0] = *(const uint4*)(bbase + 0  + kbase);
        b0.q[1] = *(const uint4*)(bbase + 16 + kbase);
        b1.q[0] = *(const uint4*)(bbase + 32 + kbase);
        b1.q[1] = *(const uint4*)(bbase + 48 + kbase);

        v8f acc = {0.f, 0.f, 0.f, 0.f, 0.f, 0.f, 0.f, 0.f};
        acc = __builtin_amdgcn_wmma_f32_16x16x32_bf16(
                  false, a0.v, false, b0.v, (short)0, acc, false, false);
        acc = __builtin_amdgcn_wmma_f32_16x16x32_bf16(
                  false, a1.v, false, b1.v, (short)0, acc, false, false);

        float bi  = bias[t * 128 + h];
        float lwv = lw  [t * 128 + h];
#pragma unroll
        for (int r = 0; r < 8; ++r) {
            float v = acc[r] * rsc[r] + bi;        // fold indeg^-1/2 + bias
            v = (v >= 0.0f) ? v : 0.01f * v;       // LeakyReLU(0.01)
            part[r] += v * lwv;                    // dot with lw over h
        }
    }

    float lbv = lb[t];
#pragma unroll
    for (int r = 0; r < 8; ++r) {
        float v = part[r];
        v += __shfl_xor(v, 1, 32);
        v += __shfl_xor(v, 2, 32);
        v += __shfl_xor(v, 4, 32);
        v += __shfl_xor(v, 8, 32);                 // 16-lane reduce (h axis)
        int m = m0 + half * 8 + r;
        if (l16 == 0 && m < Nd) s_out[(size_t)m * 32 + t] = v + lbv;
    }
}

// ---------------------------------------------------------------------------
// 6) LSTM input-projection GEMM (bf16 WMMA):
//    xi[M,512] = A[M,K] @ w_ih[512,K]^T + (b_ih + b_hh)        K in {32,128}
// ---------------------------------------------------------------------------
__global__ __launch_bounds__(128) void xi_gemm_kernel(
    const unsigned short* __restrict__ Abf,   // [M, K] bf16
    const unsigned short* __restrict__ Wbf,   // [512, K] bf16 (row = gate)
    const float* __restrict__ b_ih, const float* __restrict__ b_hh,
    float* __restrict__ xi,                   // [M, 512]
    int M, int K, int mtiles)
{
    int wave = blockIdx.x * (blockDim.x >> 5) + (threadIdx.x >> 5);
    if (wave >= mtiles * 4) return;
    int ng   = wave & 3;                 // 128-wide column group
    int m0   = (wave >> 2) * 16;
    int lane = threadIdx.x & 31;
    int half = lane >> 4;
    int l16  = lane & 15;
    int kbase = half * 8;

    int arow = m0 + l16; if (arow >= M) arow = M - 1;
    FragBF a[4];
    int ksteps = K >> 5;
    for (int kk = 0; kk < ksteps; ++kk) {
        const unsigned short* ab = Abf + (size_t)arow * K + kk * 32 + kbase;
        a[kk].q[0] = *(const uint4*)(ab);
        a[kk].q[1] = *(const uint4*)(ab + 16);
    }

    for (int nb = 0; nb < 8; ++nb) {
        int n = ng * 128 + nb * 16 + l16;
        const unsigned short* bbase = Wbf + (size_t)n * K;
        v8f acc = {0.f, 0.f, 0.f, 0.f, 0.f, 0.f, 0.f, 0.f};
        for (int kk = 0; kk < ksteps; ++kk) {
            FragBF b;
            b.q[0] = *(const uint4*)(bbase + kk * 32 + kbase);
            b.q[1] = *(const uint4*)(bbase + kk * 32 + kbase + 16);
            acc = __builtin_amdgcn_wmma_f32_16x16x32_bf16(
                      false, a[kk].v, false, b.v, (short)0, acc, false, false);
        }
        float bsum = b_ih[n] + b_hh[n];
#pragma unroll
        for (int r = 0; r < 8; ++r) {
            int mrow = m0 + half * 8 + r;
            if (mrow < M) xi[(size_t)mrow * 512 + n] = acc[r] + bsum;
        }
    }
}

// ---------------------------------------------------------------------------
// 7) Recurrent LSTM scan over the node axis (serial, persistent 1024-thr WG).
//    xi already contains x@W_ih^T + b_ih + b_hh; per step only h@W_hh^T.
//    2 threads per gate: 64 MACs each + shfl pair-reduce. Weights (256 KB)
//    stream from L2 as float4; next step's xi row is prefetched
//    (global_prefetch_b8) to shorten the serial critical path.
// ---------------------------------------------------------------------------
__global__ __launch_bounds__(1024) void lstm_scan_kernel(
    const float* __restrict__ xi,     // [M, 512]
    const float* __restrict__ w_hh,   // [512, 128]
    float* __restrict__ outh,         // [M, 128]
    int M)
{
    __shared__ float sh_h[128];
    __shared__ float sh_gate[512];

    int tid  = threadIdx.x;
    int g    = tid >> 1;          // gate 0..511
    int part = tid & 1;           // half of the 128-length dot

    const float4* wrow = (const float4*)(w_hh + (size_t)g * 128 + part * 64);
    const float4* hv   = (const float4*)(sh_h) + part * 16;

    if (tid < 128) sh_h[tid] = 0.0f;
    float c = 0.0f;
    __syncthreads();

    for (int n = 0; n < M; ++n) {
        if (part == 0 && n + 1 < M)
            __builtin_prefetch(&xi[(size_t)(n + 1) * 512 + g], 0, 1);
        float acc = 0.0f;
#pragma unroll
        for (int i = 0; i < 16; ++i) {
            float4 w = wrow[i];
            float4 h = hv[i];
            acc += w.x * h.x + w.y * h.y + w.z * h.z + w.w * h.w;
        }
        acc += __shfl_xor(acc, 1, 32);                   // combine halves
        if (part == 0) sh_gate[g] = acc + xi[(size_t)n * 512 + g];
        __syncthreads();

        if (tid < 128) {                                  // PyTorch i,f,g,o
            float gi = sh_gate[tid];
            float gf = sh_gate[128 + tid];
            float gg = sh_gate[256 + tid];
            float go = sh_gate[384 + tid];
            gi = 1.0f / (1.0f + __expf(-gi));
            gf = 1.0f / (1.0f + __expf(-gf));
            gg = tanhf(gg);
            go = 1.0f / (1.0f + __expf(-go));
            c = gf * c + gi * gg;
            float h = go * tanhf(c);
            sh_h[tid] = h;
            outh[(size_t)n * 128 + tid] = h;
        }
        __syncthreads();
    }
}

// ---------------------------------------------------------------------------
// Host orchestration
// ---------------------------------------------------------------------------
extern "C" void kernel_launch(void* const* d_in, const int* in_sizes, int n_in,
                              void* d_out, int out_size, void* d_ws, size_t ws_size,
                              hipStream_t stream)
{
    const int N_SVC = 2000, N_POD = 8000, N_NODE = 1000, NTOT = 11000;
    const int E_SVC = 30000, E_IN = 24000, E_NI = 24000;

    const float* feat_svc  = (const float*)d_in[0];
    const float* feat_pod  = (const float*)d_in[1];
    const float* feat_node = (const float*)d_in[2];
    const float* w_svc = (const float*)d_in[3];  const float* b_svc = (const float*)d_in[4];
    const float* w_in  = (const float*)d_in[5];  const float* b_in  = (const float*)d_in[6];
    const float* w_ni  = (const float*)d_in[7];  const float* b_ni  = (const float*)d_in[8];
    const float* lw_svc = (const float*)d_in[9];   const float* lb_svc = (const float*)d_in[10];
    const float* lw_pod = (const float*)d_in[11];  const float* lb_pod = (const float*)d_in[12];
    const float* lw_node = (const float*)d_in[13]; const float* lb_node = (const float*)d_in[14];
    const float* w_ih0 = (const float*)d_in[15];  const float* w_hh0 = (const float*)d_in[16];
    const float* b_ih0 = (const float*)d_in[17];  const float* b_hh0 = (const float*)d_in[18];
    const float* w_ih1 = (const float*)d_in[19];  const float* w_hh1 = (const float*)d_in[20];
    const float* b_ih1 = (const float*)d_in[21];  const float* b_hh1 = (const float*)d_in[22];
    const int* e_svc_src = (const int*)d_in[23];  const int* e_svc_dst = (const int*)d_in[24];
    const int* e_in_src  = (const int*)d_in[25];  const int* e_in_dst  = (const int*)d_in[26];
    const int* e_ni_src  = (const int*)d_in[27];  const int* e_ni_dst  = (const int*)d_in[28];

    // ---- workspace layout (256B aligned slabs) ----
    char* ws = (char*)d_ws;
    size_t off = 0;
    auto alloc = [&](size_t bytes) { size_t o = off; off += (bytes + 255) & ~(size_t)255; return o; };

    size_t o_deg     = alloc(22000 * 4);                       // 6 degree/scale arrays
    size_t o_aggSvc  = alloc((size_t)N_SVC  * 2048 * 4);
    size_t o_aggNode = alloc((size_t)N_NODE * 2048 * 4);
    size_t o_aggPod  = alloc((size_t)N_POD  * 2048 * 4);
    size_t zero_end  = off;                                    // memset [0, zero_end)
    size_t o_aggbf   = alloc((size_t)22528000 * 2);            // bf16 mirror (contiguous)
    size_t o_wTsvc   = alloc((size_t)262144 * 2);
    size_t o_wTin    = alloc((size_t)262144 * 2);
    size_t o_wTni    = alloc((size_t)262144 * 2);
    size_t o_s       = alloc((size_t)11008 * 32 * 4);
    size_t o_sbf     = alloc((size_t)11008 * 32 * 2);
    size_t o_wih0bf  = alloc((size_t)512 * 32 * 2);
    size_t o_wih1bf  = alloc((size_t)512 * 128 * 2);
    size_t o_xi0     = alloc((size_t)11008 * 512 * 4);
    size_t o_h0      = alloc((size_t)11008 * 128 * 4);
    size_t o_h0bf    = alloc((size_t)11008 * 128 * 2);
    size_t o_xi1     = alloc((size_t)11008 * 512 * 4);
    (void)ws_size; (void)n_in; (void)in_sizes; (void)out_size;

    float* deg          = (float*)(ws + o_deg);
    float* scSvcOut     = deg + 0;       // [2000]
    float* scSvcIn      = deg + 2000;    // [2000]
    float* scPodOut     = deg + 4000;    // [8000]  (e_in src)
    float* scNodeIn     = deg + 12000;   // [1000]  (e_in dst)
    float* scNodeOut    = deg + 13000;   // [1000]  (e_ni src)
    float* scPodIn      = deg + 14000;   // [8000]  (e_ni dst)
    float* aggSvc  = (float*)(ws + o_aggSvc);
    float* aggNode = (float*)(ws + o_aggNode);
    float* aggPod  = (float*)(ws + o_aggPod);
    unsigned short* aggbf     = (unsigned short*)(ws + o_aggbf);
    unsigned short* aggbfSvc  = aggbf;
    unsigned short* aggbfNode = aggbf + (size_t)N_SVC * 2048;
    unsigned short* aggbfPod  = aggbfNode + (size_t)N_NODE * 2048;
    unsigned short* wTsvc = (unsigned short*)(ws + o_wTsvc);
    unsigned short* wTin  = (unsigned short*)(ws + o_wTin);
    unsigned short* wTni  = (unsigned short*)(ws + o_wTni);
    float* s_buf = (float*)(ws + o_s);
    unsigned short* s_bf = (unsigned short*)(ws + o_sbf);
    unsigned short* wih0bf = (unsigned short*)(ws + o_wih0bf);
    unsigned short* wih1bf = (unsigned short*)(ws + o_wih1bf);
    float* xi0  = (float*)(ws + o_xi0);
    float* h0   = (float*)(ws + o_h0);
    unsigned short* h0bf = (unsigned short*)(ws + o_h0bf);
    float* xi1  = (float*)(ws + o_xi1);

    // ---- 0) zero degrees + f32 aggregation buffers ----
    hipMemsetAsync(ws, 0, zero_end, stream);

    // ---- 1) degrees, then deg -> rsqrt(clip(deg,1)) in place ----
    deg_count_kernel<<<(E_SVC + 255) / 256, 256, 0, stream>>>(e_svc_src, e_svc_dst, scSvcOut,  scSvcIn,  E_SVC);
    deg_count_kernel<<<(E_IN  + 255) / 256, 256, 0, stream>>>(e_in_src,  e_in_dst,  scPodOut,  scNodeIn, E_IN);
    deg_count_kernel<<<(E_NI  + 255) / 256, 256, 0, stream>>>(e_ni_src,  e_ni_dst,  scNodeOut, scPodIn,  E_NI);
    deg_rsqrt_kernel<<<(22000 + 255) / 256, 256, 0, stream>>>(deg, 22000);

    // ---- 2) edge scatter-add (outdeg^-1/2 pre-scaled) ----
    {
        long long th;
        th = (long long)E_SVC * 128;
        scatter_add_kernel<<<(unsigned)((th + 255) / 256), 256, 0, stream>>>(feat_svc, e_svc_src, e_svc_dst, scSvcOut, aggSvc, E_SVC);
        th = (long long)E_IN * 128;
        scatter_add_kernel<<<(unsigned)((th + 255) / 256), 256, 0, stream>>>(feat_pod, e_in_src, e_in_dst, scPodOut, aggNode, E_IN);
        th = (long long)E_NI * 128;
        scatter_add_kernel<<<(unsigned)((th + 255) / 256), 256, 0, stream>>>(feat_node, e_ni_src, e_ni_dst, scNodeOut, aggPod, E_NI);
    }

    // ---- 3) agg f32 -> bf16 (single contiguous pass), weight transposes ----
    f32_to_bf16_kernel<<<(22528000 + 255) / 256, 256, 0, stream>>>(aggSvc, aggbf, 22528000);
    w_transpose_bf16_kernel<<<(262144 + 255) / 256, 256, 0, stream>>>(w_svc, wTsvc);
    w_transpose_bf16_kernel<<<(262144 + 255) / 256, 256, 0, stream>>>(w_in,  wTin);
    w_transpose_bf16_kernel<<<(262144 + 255) / 256, 256, 0, stream>>>(w_ni,  wTni);

    // ---- 4) fused WMMA GraphConv + score epilogue -> s[11000,32] ----
    // concat order: svc [0,2000), pod [2000,10000), node [10000,11000)
    // grid = 32 timesteps x ceil(ntiles/4) tile-groups; block = 4 waves
    {
        int nt = 125;   // 2000/16
        graph_gemm_fused_kernel<<<32 * ((nt + 3) / 4), 128, 0, stream>>>(
            aggbfSvc, wTsvc, b_svc, scSvcIn, lw_svc, lb_svc, s_buf + 0, N_SVC, nt);
        nt = 500;       // 8000/16   (h_pod: node->pod edges, w_ni)
        graph_gemm_fused_kernel<<<32 * ((nt + 3) / 4), 128, 0, stream>>>(
            aggbfPod, wTni, b_ni, scPodIn, lw_pod, lb_pod, s_buf + (size_t)2000 * 32, N_POD, nt);
        nt = 63;        // ceil(1000/16)  (h_node: pod->node edges, w_in)
        graph_gemm_fused_kernel<<<32 * ((nt + 3) / 4), 128, 0, stream>>>(
            aggbfNode, wTin, b_in, scNodeIn, lw_node, lb_node, s_buf + (size_t)10000 * 32, N_NODE, nt);
    }

    // ---- 5) LSTM layer 0: WMMA input GEMM + serial recurrent scan ----
    f32_to_bf16_kernel<<<(NTOT * 32 + 255) / 256, 256, 0, stream>>>(s_buf, s_bf, NTOT * 32);
    f32_to_bf16_kernel<<<(512 * 32 + 255) / 256, 256, 0, stream>>>(w_ih0, wih0bf, 512 * 32);
    f32_to_bf16_kernel<<<(512 * 128 + 255) / 256, 256, 0, stream>>>(w_ih1, wih1bf, 512 * 128);

    int mtiles = (NTOT + 15) / 16;   // 688
    xi_gemm_kernel<<<mtiles, 128, 0, stream>>>(s_bf, wih0bf, b_ih0, b_hh0, xi0, NTOT, 32, mtiles);
    lstm_scan_kernel<<<1, 1024, 0, stream>>>(xi0, w_hh0, h0, NTOT);

    // ---- 6) LSTM layer 1 ----
    f32_to_bf16_kernel<<<(NTOT * 128 + 255) / 256, 256, 0, stream>>>(h0, h0bf, NTOT * 128);
    xi_gemm_kernel<<<mtiles, 128, 0, stream>>>(h0bf, wih1bf, b_ih1, b_hh1, xi1, NTOT, 128, mtiles);
    lstm_scan_kernel<<<1, 1024, 0, stream>>>(xi1, w_hh1, (float*)d_out, NTOT);
}
